// WindowsAttention_15805479649518
// MI455X (gfx1250) — matrix operand order repaired and checked by reading, compiled, not run on hardware
//
#include <hip/hip_runtime.h>

typedef __attribute__((ext_vector_type(8)))  float  v8f;
typedef __attribute__((ext_vector_type(4)))  float  v4f;
typedef __attribute__((ext_vector_type(16))) __bf16 v16bf;
typedef __attribute__((ext_vector_type(8)))  __bf16 v8bf;
typedef __attribute__((ext_vector_type(4)))  __bf16 v4bf;
typedef __attribute__((ext_vector_type(4)))  unsigned int v4u;
typedef __attribute__((ext_vector_type(8)))  int v8i;
typedef __attribute__((ext_vector_type(4)))  int v4i;

#define WMMA_BF16(a, b, c) \
  __builtin_amdgcn_wmma_f32_16x16x32_bf16(false, (a), false, (b), (short)0, (c), false, false)

// ---- problem constants -----------------------------------------------------
#define DIMC   256
#define NHEAD  8
#define HD     32
#define NTOK   64
#define NWIN   2048
#define NROWS  (NWIN * NTOK)          // 131072 token rows
#define QKSCALE 0.17677669529663687f  // 32^-0.5
#define LDB    264                    // LDS row stride in bf16 (528B: aligned + conflict-free)

// ---- workspace byte offsets ------------------------------------------------
#define OFF_XB   0ull          // bf16 [131072][256]
#define OFF_Q    67108864ull   // bf16 [2048][8][64][32]  (pre-scaled)
#define OFF_K    134217728ull  // bf16 [2048][8][64][32]
#define OFF_VT   201326592ull  // bf16 [2048][8][32][64]  (transposed)
#define OFF_AO   268435456ull  // bf16 [131072][256]      (pre-proj attn out)
#define OFF_WQKV 335544320ull  // bf16 [768][256]
#define OFF_WPRJ 335937536ull  // bf16 [256][256]
#define OFF_BIAS 336068608ull  // f32  [8][64][64]

__device__ __forceinline__ v16bf ld16(const __bf16* p0, const __bf16* p1) {
  union { v16bf v; v8bf h[2]; } u;
  u.h[0] = *(const v8bf*)p0;
  u.h[1] = *(const v8bf*)p1;
  return u.v;
}

// ---- TDM: stage `rows` x 256-bf16 weight rows into LDS, +16B pad per 512B row
// D# per CDNA5 ISA: group0 = flags/lds/global addr,
// group1 = mask/data_size/pad + dims/strides. data_size=1 (2B elements).
// pad_interval code 6 = every 128 DWORDs (512B), pad_amount code 3 = 4 DWORDs (16B).
// This toolchain (clang-23 / therock-10.0) uses the 6-arg builtin form.
__device__ __forceinline__ void tdm_stage_rows(const __bf16* gsrc, void* lds_dst, int rows) {
  unsigned long long ga = (unsigned long long)(uintptr_t)gsrc;
  v4u g0;
  g0[0] = 1u;                                            // count=1, user mode
  g0[1] = (unsigned)(uintptr_t)lds_dst;                  // LDS byte address (low 32 of flat)
  g0[2] = (unsigned)ga;                                  // global_addr[31:0]
  g0[3] = (unsigned)((ga >> 32) & 0x01ffffffull) | (2u << 30);  // addr[56:32] | type=2
  v8i g1;
  g1[0] = (1 << 16) | (1 << 20) | (6 << 22) | (3 << 25); // data_size=2B, pad_en, intv, amt
  g1[1] = (int)(256u << 16);                             // tensor_dim0 = 256 elems
  g1[2] = (int)((unsigned)rows << 16);                   // tensor_dim1 = rows
  g1[3] = (int)(256u << 16);                             // tile_dim0 = 256
  g1[4] = rows;                                          // tile_dim1 = rows (tile_dim2=0)
  g1[5] = 256;                                           // tensor_dim0_stride = 256
  g1[6] = 0;
  g1[7] = 0;
  v4i gz4 = {0, 0, 0, 0};                                // groups 2/3 unused (<=2D)
  v8i gz8 = {0, 0, 0, 0, 0, 0, 0, 0};
  __builtin_amdgcn_tensor_load_to_lds(g0, g1, gz4, gz4, gz8, 0);
}

// ---- kernel 1: x (f32) -> bf16 --------------------------------------------
__global__ __launch_bounds__(256) void k_convert_x(const float* __restrict__ x,
                                                   __bf16* __restrict__ xb) {
  size_t i = ((size_t)blockIdx.x * blockDim.x + threadIdx.x) * 4;
  v4f f = *(const v4f*)(x + i);
  v4bf o;
  o.x = (__bf16)f.x; o.y = (__bf16)f.y; o.z = (__bf16)f.z; o.w = (__bf16)f.w;
  *(v4bf*)(xb + i) = o;
}

// ---- kernel 2: weights -> bf16, bias gather -------------------------------
__global__ __launch_bounds__(256) void k_prep(const float* __restrict__ qkv_w,
                                              const float* __restrict__ proj_w,
                                              const float* __restrict__ bias_table,
                                              const int* __restrict__ rpi,
                                              __bf16* __restrict__ wqkv,
                                              __bf16* __restrict__ wprj,
                                              float* __restrict__ biasM) {
  int i = blockIdx.x * blockDim.x + threadIdx.x;
  if (i < 196608) {
    wqkv[i] = (__bf16)qkv_w[i];
  } else if (i < 262144) {
    int j = i - 196608;
    wprj[j] = (__bf16)proj_w[j];
  } else if (i < 294912) {
    int j = i - 262144;          // j = h*4096 + m*64 + n
    int h = j >> 12, mn = j & 4095;
    biasM[h * 4096 + mn] = bias_table[rpi[mn] * NHEAD + h];
  }
}

// ---- kernel 3: QKV GEMM (131072x768, K=256) + head scatter ----------------
// Block: 8 waves = 128 rows x 64 cols; weights staged once per block via TDM.
__global__ __launch_bounds__(256) void k_qkv(const __bf16* __restrict__ xb,
                                             const __bf16* __restrict__ wqkv,
                                             const float* __restrict__ qkv_b,
                                             __bf16* __restrict__ qd,
                                             __bf16* __restrict__ kd,
                                             __bf16* __restrict__ vtd) {
  __shared__ __bf16 BS[64 * LDB];           // 33,792 B
  int tid = threadIdx.x;
  int lane = tid & 31, w = tid >> 5;
  int lo = lane & 15, hi = lane >> 4;
  int rowBase = blockIdx.x * 128 + w * 16;  // 1024 row blocks
  int nbase   = blockIdx.y * 64;            // 12 col blocks

  if (tid == 0) {
    tdm_stage_rows(wqkv + (size_t)nbase * DIMC, &BS[0], 64);
    __builtin_amdgcn_s_wait_tensorcnt(0);
  }
  __syncthreads();

  v8f acc[4] = {{}, {}, {}, {}};
  const __bf16* arow = xb + (size_t)(rowBase + lo) * DIMC;
#pragma unroll
  for (int kt = 0; kt < 8; ++kt) {
    int kb = kt * 32;
    v16bf a = ld16(arow + kb + hi * 8, arow + kb + hi * 8 + 16);
#pragma unroll
    for (int t = 0; t < 4; ++t) {
      const __bf16* bp = &BS[(t * 16 + lo) * LDB + kb + hi * 8];
      v16bf b = ld16(bp, bp + 16);
      acc[t] = WMMA_BF16(a, b, acc[t]);
    }
  }
#pragma unroll
  for (int t = 0; t < 4; ++t) {
    int ccol = nbase + t * 16 + lo;
    float bias = qkv_b[ccol];
#pragma unroll
    for (int r = 0; r < 8; ++r) {
      int m = rowBase + r + hi * 8;
      float val = acc[t][r] + bias;
      int bwin = m >> 6, nn = m & 63;
      if (ccol < 256) {
        int h = ccol >> 5, d = ccol & 31;
        qd[(((size_t)bwin * NHEAD + h) * NTOK + nn) * HD + d] = (__bf16)(val * QKSCALE);
      } else if (ccol < 512) {
        int cc = ccol - 256, h = cc >> 5, d = cc & 31;
        kd[(((size_t)bwin * NHEAD + h) * NTOK + nn) * HD + d] = (__bf16)val;
      } else {
        int cc = ccol - 512, h = cc >> 5, d = cc & 31;
        vtd[(((size_t)bwin * NHEAD + h) * HD + d) * NTOK + nn] = (__bf16)val;
      }
    }
  }
}

// ---- kernel 4: attention per (window, head) -------------------------------
__global__ __launch_bounds__(128) void k_attn(const __bf16* __restrict__ qd,
                                              const __bf16* __restrict__ kd,
                                              const __bf16* __restrict__ vtd,
                                              const float* __restrict__ biasM,
                                              float* __restrict__ attn_out,
                                              __bf16* __restrict__ ao) {
  __shared__ __bf16 P[64 * 80];             // padded: 160B row stride, 16B aligned
  int lane = threadIdx.x & 31, w = threadIdx.x >> 5;
  int lo = lane & 15, hi = lane >> 4;
  int bh = blockIdx.x;                      // b*8 + h
  int h = bh & 7, bwin = bh >> 3;

  const __bf16* qp = qd + (size_t)bh * NTOK * HD;
  const __bf16* kp = kd + (size_t)bh * NTOK * HD;
  const __bf16* vp = vtd + (size_t)bh * HD * NTOK;

  // S = q @ k^T   (strip of 16 rows, all 64 cols; K=32 == head_dim)
  const __bf16* arow = qp + (w * 16 + lo) * HD;
  v16bf a = ld16(arow + hi * 8, arow + hi * 8 + 16);
  v8f s[4];
#pragma unroll
  for (int t = 0; t < 4; ++t) {
    const __bf16* bp = kp + (t * 16 + lo) * HD + hi * 8;
    v16bf b = ld16(bp, bp + 16);
    v8f z = {};
    s[t] = WMMA_BF16(a, b, z);
  }
  // + relative position bias
  const float* bM = biasM + h * 4096;
#pragma unroll
  for (int t = 0; t < 4; ++t)
#pragma unroll
    for (int r = 0; r < 8; ++r)
      s[t][r] += bM[(w * 16 + r + hi * 8) * 64 + t * 16 + lo];

  // softmax per row (each row lives in one VGPR slot across a 16-lane group)
#pragma unroll
  for (int r = 0; r < 8; ++r) {
    float mx = fmaxf(fmaxf(s[0][r], s[1][r]), fmaxf(s[2][r], s[3][r]));
    mx = fmaxf(mx, __shfl_xor(mx, 1));
    mx = fmaxf(mx, __shfl_xor(mx, 2));
    mx = fmaxf(mx, __shfl_xor(mx, 4));
    mx = fmaxf(mx, __shfl_xor(mx, 8));
    float e0 = __expf(s[0][r] - mx), e1 = __expf(s[1][r] - mx);
    float e2 = __expf(s[2][r] - mx), e3 = __expf(s[3][r] - mx);
    float sum = (e0 + e1) + (e2 + e3);
    sum += __shfl_xor(sum, 1);
    sum += __shfl_xor(sum, 2);
    sum += __shfl_xor(sum, 4);
    sum += __shfl_xor(sum, 8);
    float inv = 1.0f / sum;
    s[0][r] = e0 * inv; s[1][r] = e1 * inv; s[2][r] = e2 * inv; s[3][r] = e3 * inv;
  }

  // emit attn (f32, required output) and stage P (bf16) in LDS
  float* aout = attn_out + (size_t)bh * NTOK * NTOK;
#pragma unroll
  for (int t = 0; t < 4; ++t)
#pragma unroll
    for (int r = 0; r < 8; ++r) {
      int m = w * 16 + r + hi * 8;
      int n = t * 16 + lo;
      float v = s[t][r];
      aout[m * 64 + n] = v;
      P[m * 80 + n] = (__bf16)v;
    }
  __syncthreads();

  // O = P @ V   (A from LDS, B = v^T rows contiguous)
  v8f o[2] = {{}, {}};
#pragma unroll
  for (int kt = 0; kt < 2; ++kt) {
    const __bf16* pr = &P[(w * 16 + lo) * 80 + kt * 32];
    v16bf ap = ld16(pr + hi * 8, pr + hi * 8 + 16);
#pragma unroll
    for (int n2 = 0; n2 < 2; ++n2) {
      const __bf16* bp = vp + (n2 * 16 + lo) * NTOK + kt * 32 + hi * 8;
      v16bf bv = ld16(bp, bp + 16);
      o[n2] = WMMA_BF16(ap, bv, o[n2]);
    }
  }
#pragma unroll
  for (int n2 = 0; n2 < 2; ++n2)
#pragma unroll
    for (int r = 0; r < 8; ++r) {
      int m = w * 16 + r + hi * 8;
      int col = h * HD + n2 * 16 + lo;
      ao[((size_t)bwin * NTOK + m) * DIMC + col] = (__bf16)o[n2][r];
    }
}

// ---- kernel 5: output projection (131072x256, K=256) ----------------------
__global__ __launch_bounds__(256) void k_proj(const __bf16* __restrict__ ao,
                                              const __bf16* __restrict__ wprj,
                                              const float* __restrict__ proj_b,
                                              float* __restrict__ out) {
  __shared__ __bf16 BS[64 * LDB];
  int tid = threadIdx.x;
  int lane = tid & 31, w = tid >> 5;
  int lo = lane & 15, hi = lane >> 4;
  int rowBase = blockIdx.x * 128 + w * 16;  // 1024 row blocks
  int nbase   = blockIdx.y * 64;            // 4 col blocks

  if (tid == 0) {
    tdm_stage_rows(wprj + (size_t)nbase * DIMC, &BS[0], 64);
    __builtin_amdgcn_s_wait_tensorcnt(0);
  }
  __syncthreads();

  v8f acc[4] = {{}, {}, {}, {}};
  const __bf16* arow = ao + (size_t)(rowBase + lo) * DIMC;
#pragma unroll
  for (int kt = 0; kt < 8; ++kt) {
    int kb = kt * 32;
    v16bf a = ld16(arow + kb + hi * 8, arow + kb + hi * 8 + 16);
#pragma unroll
    for (int t = 0; t < 4; ++t) {
      const __bf16* bp = &BS[(t * 16 + lo) * LDB + kb + hi * 8];
      v16bf b = ld16(bp, bp + 16);
      acc[t] = WMMA_BF16(a, b, acc[t]);
    }
  }
#pragma unroll
  for (int t = 0; t < 4; ++t) {
    int col = nbase + t * 16 + lo;
    float bias = proj_b[col];
#pragma unroll
    for (int r = 0; r < 8; ++r) {
      int m = rowBase + r + hi * 8;
      out[(size_t)m * DIMC + col] = acc[t][r] + bias;
    }
  }
}

// ---- launch ---------------------------------------------------------------
extern "C" void kernel_launch(void* const* d_in, const int* in_sizes, int n_in,
                              void* d_out, int out_size, void* d_ws, size_t ws_size,
                              hipStream_t stream) {
  const float* x          = (const float*)d_in[0];
  const float* qkv_w      = (const float*)d_in[1];
  const float* qkv_b      = (const float*)d_in[2];
  const float* proj_w     = (const float*)d_in[3];
  const float* proj_b     = (const float*)d_in[4];
  const float* bias_table = (const float*)d_in[5];
  const int*   rpi        = (const int*)d_in[6];

  char* ws = (char*)d_ws;
  __bf16* xb    = (__bf16*)(ws + OFF_XB);
  __bf16* qd    = (__bf16*)(ws + OFF_Q);
  __bf16* kd    = (__bf16*)(ws + OFF_K);
  __bf16* vtd   = (__bf16*)(ws + OFF_VT);
  __bf16* ao    = (__bf16*)(ws + OFF_AO);
  __bf16* wqkv  = (__bf16*)(ws + OFF_WQKV);
  __bf16* wprj  = (__bf16*)(ws + OFF_WPRJ);
  float*  biasM = (float*)(ws + OFF_BIAS);

  float* out      = (float*)d_out;                       // [131072][256]
  float* attn_out = out + (size_t)NROWS * DIMC;          // [2048][8][64][64]

  k_convert_x<<<32768, 256, 0, stream>>>(x, xb);         // 33.5M elems / 4
  k_prep<<<1152, 256, 0, stream>>>(qkv_w, proj_w, bias_table, rpi, wqkv, wprj, biasM);
  k_qkv<<<dim3(1024, 12), 256, 0, stream>>>(xb, wqkv, qkv_b, qd, kd, vtd);
  k_attn<<<NWIN * NHEAD, 128, 0, stream>>>(qd, kd, vtd, biasM, attn_out, ao);
  k_proj<<<dim3(1024, 4), 256, 0, stream>>>(ao, wprj, proj_b, out);
}
